// Codebook_EMA_37306085933615
// MI455X (gfx1250) — matrix operand order repaired and checked by reading, not compiled, not run on hardware
//
#include <hip/hip_runtime.h>
#include <stdint.h>
#include <math.h>

// ---------------------------------------------------------------------------
// VQ-VAE codebook quantization for MI455X (gfx1250, wave32, WMMA)
//   z:         [8, 256, 16,16,16] f32   (N = 8*4096 = 32768 latent vectors)
//   embedding: [1024, 256] f32
// Outputs (flat concat): z_q [8,256,16,16,16] | loss | perplexity |
//                        encodings [32768,1024] | min_idx [32768,1]
// ---------------------------------------------------------------------------

typedef __attribute__((ext_vector_type(16))) __bf16 v16bf;
typedef __attribute__((ext_vector_type(8)))  float  v8f;

struct __align__(16) U4 { unsigned int x, y, z, w; };

#define NVEC  32768
#define LAT   256
#define CODES 1024
#define SPAT  4096          // 16*16*16 spatial positions per batch
#define BSTRIDE 1048576     // 256*4096 floats per batch in z
#define BROW  264           // padded LDS row stride (ushorts) -> 132 dwords
#define BUFSZ (16 * BROW)   // one staged code tile (16 codes x 256 ch, padded)

__device__ __forceinline__ unsigned short f2b(float f) {
  union { float f; unsigned int u; } c; c.f = f;
  unsigned int u = c.u;
  return (unsigned short)((u + 0x7FFFu + ((u >> 16) & 1u)) >> 16); // RNE
}

// --- K1: codebook -> bf16, plus ||e||^2 per code --------------------------
__global__ void prep_embed(const float* __restrict__ emb,
                           unsigned short* __restrict__ ebf,
                           float* __restrict__ enorm) {
  int k = blockIdx.x, c = threadIdx.x;
  float v = emb[k * LAT + c];
  ebf[k * LAT + c] = f2b(v);
  __shared__ float red[256];
  red[c] = v * v;
  __syncthreads();
  for (int s = 128; s > 0; s >>= 1) {
    if (c < s) red[c] += red[c + s];
    __syncthreads();
  }
  if (c == 0) enorm[k] = red[0];
}

// --- K2: z [B,C,S] -> z_flat [B*S, C] in bf16 via LDS tile transpose ------
__global__ void prep_z(const float* __restrict__ z,
                       unsigned short* __restrict__ zbf) {
  __shared__ float tile[32][33];
  int tx = threadIdx.x & 31, ty = threadIdx.x >> 5;
  int st = blockIdx.x, ct = blockIdx.y, b = blockIdx.z;
  const float* zp = z + (size_t)b * BSTRIDE;
#pragma unroll
  for (int i = 0; i < 4; ++i) {
    int cl = ty + i * 8;                       // local c
    tile[cl][tx] = zp[(size_t)(ct * 32 + cl) * SPAT + st * 32 + tx]; // coalesced (s)
  }
  __syncthreads();
#pragma unroll
  for (int i = 0; i < 4; ++i) {
    int sl = ty + i * 8;                       // local s
    zbf[((size_t)(b * SPAT + st * 32 + sl)) * LAT + ct * 32 + tx] =
        f2b(tile[tx][sl]);                     // coalesced (c)
  }
}

// --- K3: fused distance GEMM (bf16 WMMA) + argmin + gather + loss ---------
__global__ void __launch_bounds__(256) vq_main(
    const unsigned short* __restrict__ zbf,
    const unsigned short* __restrict__ ebf,
    const float* __restrict__ enorm,
    const float* __restrict__ z,
    const float* __restrict__ emb,
    float* __restrict__ zq_out,
    float* __restrict__ enc_out,
    float* __restrict__ idx_out,
    float* __restrict__ counts,
    float* __restrict__ sse_acc) {
  __shared__ unsigned short btile[2][BUFSZ];   // double-buffered B tile (~17 KB)

  const int lane = threadIdx.x & 31;
  const int wave = threadIdx.x >> 5;
  const int gw   = blockIdx.x * 8 + wave;      // 2048 waves, 16 rows each
  const int m0   = gw * 16;
  const int m    = lane & 15;
  const int hi   = lane >> 4;

  // A fragments for all 256 channels (ISA 16-bit A 16x32 layout):
  //   lanes 0-15 hold K = {0..7, 16..23}, lanes 16-31 hold K = {8..15, 24..31}
  const unsigned short* za = zbf + (size_t)(m0 + m) * LAT;
  v16bf a[8];
#pragma unroll
  for (int ck = 0; ck < 8; ++ck) {
    union { U4 u[2]; v16bf v; } pk;
    pk.u[0] = *(const U4*)(za + ck * 32 + hi * 8);        // K = hi*8 + 0..7
    pk.u[1] = *(const U4*)(za + ck * 32 + 16 + hi * 8);   // K = 16 + hi*8 + 0..7
    a[ck] = pk.v;
  }

  // Cooperative B staging: thread -> (code row, 32-byte segment) of the tile
  const int fc = threadIdx.x >> 4;             // 0..15 code within tile
  const int fs = threadIdx.x & 15;             // 32-byte segment within row
  const unsigned short* gsrc = ebf + (size_t)fc * LAT + fs * 16;
  U4 ga = *(const U4*)(gsrc);
  U4 gb = *(const U4*)(gsrc + 8);

  // Per-lane running argmin (this lane's column only): no cross-lane traffic
  // inside the hot loop -> pure v_cmp + v_cndmask epilogue per tile.
  float bestv[8];
  int   bidx[8];
#pragma unroll
  for (int r = 0; r < 8; ++r) { bestv[r] = 3.4e38f; bidx[r] = 0; }

  for (int t = 0; t < 64; ++t) {               // 64 tiles of 16 codes
    unsigned short* bt = &btile[t & 1][0];
    *(U4*)(bt + fc * BROW + fs * 16)     = ga;
    *(U4*)(bt + fc * BROW + fs * 16 + 8) = gb;
    __syncthreads();
    if (t + 1 < 64) {                          // overlap next tile's fetch
      const unsigned short* ns = gsrc + (size_t)(t + 1) * 16 * LAT;
      ga = *(const U4*)(ns);
      gb = *(const U4*)(ns + 8);
    }

    // B fragment (Kx16): lane = column n, lanes 0-15 hold K={0..7,16..23}, 16-31 K={8..15,24..31}
    const unsigned short* eb = bt + m * BROW + hi * 8;
    v8f acc = {};
#pragma unroll
    for (int ck = 0; ck < 8; ++ck) {
      union { U4 u[2]; v16bf v; } pk;
      pk.u[0] = *(const U4*)(eb + ck * 32);
      pk.u[1] = *(const U4*)(eb + ck * 32 + 16);
      acc = __builtin_amdgcn_wmma_f32_16x16x32_bf16(
          false, a[ck], false, pk.v, (short)0, acc, false, false);
    }

    // dist(row, n) = ||e_n||^2 - 2 * <z_row, e_n>   (||z||^2 constant per row)
    const int   code0 = t * 16;
    const float en  = enorm[code0 + m];        // n = lane&15
    const int   idx = code0 + m;
#pragma unroll
    for (int r = 0; r < 8; ++r) {              // row = r + hi*8
      float v = __builtin_fmaf(-2.0f, acc[r], en);
      bool tk = v < bestv[r];                  // strict <: first code wins ties
      bestv[r] = tk ? v : bestv[r];
      bidx[r]  = tk ? idx : bidx[r];
    }
    __syncthreads();
  }

  // One cross-lane min+argmin per row over the 16-lane half (32 steps total).
#pragma unroll
  for (int r = 0; r < 8; ++r) {
    float v  = bestv[r];
    int   id = bidx[r];
#pragma unroll
    for (int off = 1; off < 16; off <<= 1) {
      float ov = __shfl_xor(v, off);
      int   oi = __shfl_xor(id, off);
      bool tk = (ov < v) || (ov == v && oi < id);
      v  = tk ? ov : v;
      id = tk ? oi : id;
    }
    bestv[r] = v;
    bidx[r]  = id;
  }

  // Writers: lane 0 owns rows m0+0..7, lane 16 owns rows m0+8..15
  if (m == 0) {
#pragma unroll
    for (int r = 0; r < 8; ++r) {
      int row = m0 + hi * 8 + r;
      int idx = bidx[r];
      idx_out[row] = (float)idx;
      enc_out[(size_t)row * CODES + idx] = 1.0f;
      atomicAdd(&counts[idx], 1.0f);
    }
  }

  // Each lane takes ownership of row (lane&15) for the gather/store phase.
  int myIdx = __shfl(bidx[m & 7], (m & 8) ? 16 : 0);

  const int   bb = m0 >> 12;                   // batch index (m0 multiple of 16)
  const int   s0 = m0 & (SPAT - 1);
  const float* erow = emb + (size_t)myIdx * LAT;  // exact f32 codebook row (L2-hot)
  float sse = 0.f;
  for (int cc = 0; cc < LAT; cc += 2) {
    int c = cc + hi;
    float q = erow[c];
    size_t off = (size_t)bb * BSTRIDE + (size_t)c * SPAT + (s0 + m); // 16 consec lanes
    float zv = z[off];
    zq_out[off] = q;                           // straight-through output == z_q
    float d = q - zv;
    sse += d * d;
  }
#pragma unroll
  for (int off = 1; off < 32; off <<= 1) sse += __shfl_xor(sse, off);
  if (lane == 0) atomicAdd(sse_acc, sse);
}

// --- K4: perplexity + loss scalars ----------------------------------------
__global__ void finalize(const float* __restrict__ counts,
                         const float* __restrict__ sse_acc,
                         float* __restrict__ loss_out,
                         float* __restrict__ ppl_out) {
  __shared__ float red[1024];
  int t = threadIdx.x;
  float p = counts[t] * (1.0f / (float)NVEC);
  red[t] = p * logf(p + 1e-10f);
  __syncthreads();
  for (int s = 512; s > 0; s >>= 1) {
    if (t < s) red[t] += red[t + s];
    __syncthreads();
  }
  if (t == 0) {
    *ppl_out  = expf(-red[0]);
    *loss_out = 0.25f * sse_acc[0] / 8388608.0f; // mean over N*LAT elements
  }
}

// ---------------------------------------------------------------------------
extern "C" void kernel_launch(void* const* d_in, const int* in_sizes, int n_in,
                              void* d_out, int out_size, void* d_ws, size_t ws_size,
                              hipStream_t stream) {
  const float* z   = (const float*)d_in[0];  // 8*256*4096 f32
  const float* emb = (const float*)d_in[1];  // 1024*256 f32

  char* ws = (char*)d_ws;
  float*          enorm  = (float*)(ws + 0);              // 4 KB
  float*          counts = (float*)(ws + 4096);           // 4 KB
  float*          sse    = (float*)(ws + 8192);           // 16 B
  unsigned short* ebf    = (unsigned short*)(ws + 16384); // 512 KB
  unsigned short* zbf    = (unsigned short*)(ws + 16384 + 524288); // 16 MB

  float* out  = (float*)d_out;
  float* zq   = out;                                   // 8,388,608
  float* loss = out + 8388608;                         // 1
  float* ppl  = out + 8388609;                         // 1
  float* enc  = out + 8388610;                         // 33,554,432
  float* midx = out + 8388610 + 33554432;              // 32,768

  // zero accumulators and the one-hot encodings region (graph-capture safe)
  hipMemsetAsync(counts, 0, 4096 + 64, stream);
  hipMemsetAsync(enc, 0, (size_t)33554432 * sizeof(float), stream);

  prep_embed<<<CODES, 256, 0, stream>>>(emb, ebf, enorm);
  prep_z<<<dim3(SPAT / 32, LAT / 32, 8), 256, 0, stream>>>(z, zbf);
  vq_main<<<NVEC / (16 * 8), 256, 0, stream>>>(zbf, ebf, enorm, z, emb,
                                               zq, enc, midx, counts, sse);
  finalize<<<1, 1024, 0, stream>>>(counts, sse, loss, ppl);
}
